// GNN_60627758350590
// MI455X (gfx1250) — compile-verified, matrix operand order checked by reading, not run on hardware
//
#include <hip/hip_runtime.h>
#include <hip/hip_fp16.h>

typedef __attribute__((ext_vector_type(16))) _Float16 v16h;
typedef __attribute__((ext_vector_type(8)))  _Float16 v8h;
typedef __attribute__((ext_vector_type(4)))  _Float16 v4h;
typedef __attribute__((ext_vector_type(8)))  float    v8f;

#define N_NODES 100000
#define IN_F    384
#define HID     128
#define OUTF    64
#define NCLS    2
#define E_EDGES 1600000
#define LN_EPS  1e-5f
#define SLOPE   0.01f

// WMMA f16 16x16x32 fragment (CDNA5 ISA 7.12.2): for lane half hi,
// halves 0..7  = K [hi*8,  hi*8+8)   and halves 8..15 = K [16+hi*8, +8)
// -> a fragment is two contiguous 16-byte chunks of a row-major f16 row.
union frag16 { v16h v; v8h h[2]; };

// ============================================================
// f32 -> f16 elementwise (weights pre-conversion, run once per call)
// ============================================================
__global__ __launch_bounds__(256) void k_cvt(const float* __restrict__ s,
                                             _Float16* __restrict__ d, int n)
{
    int i = blockIdx.x * 256 + threadIdx.x;
    if (i < n) d[i] = (_Float16)s[i];
}

// ============================================================
// Fused LayerNorm(384) + fc1 GEMM -> h0 [N,128] f16
// One wave per 32-node tile: each lane owns one row for LN (one-pass
// sum/sumsq), then dual-tile WMMA so every B fragment feeds 2 WMMAs
// (halves weight L2 traffic vs 16-row tiles).
// ============================================================
__global__ __launch_bounds__(32) void k_ln_fc1(
    const float* __restrict__ x, const float* __restrict__ g,
    const float* __restrict__ bln, const _Float16* __restrict__ wh,
    const float* __restrict__ bias, _Float16* __restrict__ outh)
{
    __shared__ _Float16 xh[32][IN_F];            // 24 KB

    const int lane = threadIdx.x;
    const int tile = blockIdx.x;                 // 32-row tile
    const int r = lane & 15, hi = lane >> 4;

    // ---- LayerNorm: one row per lane, single streaming pass ----
    const float4* xr = (const float4*)(x + (long)(tile * 32 + lane) * IN_F);
    float s = 0.f, ss = 0.f;
    for (int q = 0; q < IN_F / 4; ++q) {
        float4 v = xr[q];
        s  += v.x + v.y + v.z + v.w;
        ss += v.x * v.x + v.y * v.y + v.z * v.z + v.w * v.w;
    }
    const float m  = s * (1.0f / IN_F);
    const float var = ss * (1.0f / IN_F) - m * m;
    const float rs = rsqrtf(var + LN_EPS);
    const float4* gr = (const float4*)g;
    const float4* br = (const float4*)bln;
    for (int q = 0; q < IN_F / 4; ++q) {         // re-read row (L1 hit)
        float4 v = xr[q], gg = gr[q], bb = br[q];
        _Float16* d = &xh[lane][q * 4];
        d[0] = (_Float16)((v.x - m) * rs * gg.x + bb.x);
        d[1] = (_Float16)((v.y - m) * rs * gg.y + bb.y);
        d[2] = (_Float16)((v.z - m) * rs * gg.z + bb.z);
        d[3] = (_Float16)((v.w - m) * rs * gg.w + bb.w);
    }
    __syncthreads();

    // ---- GEMM: 12 K-steps x 8 N-tiles, 2 row-tiles share each B frag ----
    v8f acc0[HID / 16] = {}, acc1[HID / 16] = {};
    for (int kk = 0; kk < IN_F / 32; ++kk) {
        frag16 a0, a1;                            // ds_load_b128 x4
        a0.h[0] = *(const v8h*)&xh[r][kk * 32 + hi * 8];
        a0.h[1] = *(const v8h*)&xh[r][kk * 32 + 16 + hi * 8];
        a1.h[0] = *(const v8h*)&xh[16 + r][kk * 32 + hi * 8];
        a1.h[1] = *(const v8h*)&xh[16 + r][kk * 32 + 16 + hi * 8];
        #pragma unroll
        for (int t = 0; t < HID / 16; ++t) {
            const _Float16* wp = wh + (long)(t * 16 + r) * IN_F + kk * 32;
            frag16 b;                             // global_load_b128 x2
            b.h[0] = *(const v8h*)(wp + hi * 8);
            b.h[1] = *(const v8h*)(wp + 16 + hi * 8);
            acc0[t] = __builtin_amdgcn_wmma_f32_16x16x32_f16(
                false, a0.v, false, b.v, (short)0, acc0[t], false, false);
            acc1[t] = __builtin_amdgcn_wmma_f32_16x16x32_f16(
                false, a1.v, false, b.v, (short)0, acc1[t], false, false);
        }
    }

    #pragma unroll
    for (int t = 0; t < HID / 16; ++t) {
        const int n = t * 16 + r;
        const float bn = bias[n];
        #pragma unroll
        for (int i = 0; i < 8; ++i) {            // D: VGPR i -> row i+8*hi
            int row0 = tile * 32 + i + 8 * hi;
            outh[(long)row0 * HID + n]        = (_Float16)(acc0[t][i] + bn);
            outh[(long)(row0 + 16) * HID + n] = (_Float16)(acc1[t][i] + bn);
        }
    }
}

// ============================================================
// Zero the f32 aggregation buffer (re-init every call -> deterministic)
// ============================================================
__global__ __launch_bounds__(256) void k_zero(float* __restrict__ p, long n)
{
    long i = (long)blockIdx.x * blockDim.x + threadIdx.x;
    if (i < n) p[i] = 0.f;
}

// ============================================================
// Scatter-add: agg[dst] += w_e * h[src]; 32 lanes/edge, each lane moves
// 4 halves (one b64 gather, L2-resident) -> 4 global_atomic_add_f32.
// ============================================================
__global__ __launch_bounds__(256) void k_scatter(
    const int* __restrict__ ei, const float* __restrict__ ew,
    const _Float16* __restrict__ h, float* __restrict__ agg)
{
    long gid = (long)blockIdx.x * blockDim.x + threadIdx.x;
    long e   = gid >> 5;
    int lane = (int)(gid & 31);
    if (e >= E_EDGES) return;

    const int   src = ei[e];
    const int   dst = ei[E_EDGES + e];
    const float we  = ew[e];

    const v4h hv = *(const v4h*)(h + (long)src * HID + lane * 4);
    float* ag = agg + (long)dst * HID + lane * 4;
    atomicAdd(ag + 0, (float)hv[0] * we);
    atomicAdd(ag + 1, (float)hv[1] * we);
    atomicAdd(ag + 2, (float)hv[2] * we);
    atomicAdd(ag + 3, (float)hv[3] * we);
}

// ============================================================
// GraphConv linear: out = leaky( agg@wrel^T + brel + h@wroot^T )
// One wave per 32-row tile; both passes accumulate into the same WMMA
// accumulators; each B fragment feeds 2 WMMAs.
// OT = _Float16 (conv1) or float (conv2, feeds LN tail).
// ============================================================
template <int OUTC, typename OT>
__global__ __launch_bounds__(32) void k_conv(
    const float* __restrict__ agg, const _Float16* __restrict__ hroot,
    const _Float16* __restrict__ wrel, const float* __restrict__ brel,
    const _Float16* __restrict__ wroot, OT* __restrict__ out)
{
    const int lane = threadIdx.x;
    const int tile = blockIdx.x;                 // 32-row tile
    const int r = lane & 15, hi = lane >> 4;

    v8f acc0[OUTC / 16] = {}, acc1[OUTC / 16] = {};

    // ---- pass 0: rel term, A from f32 agg (cvt_pk in regs) ----
    for (int kk = 0; kk < HID / 32; ++kk) {
        frag16 a0, a1;
        #pragma unroll
        for (int mrow = 0; mrow < 2; ++mrow) {
            const float* ap =
                agg + (long)(tile * 32 + mrow * 16 + r) * HID + kk * 32;
            float4 f0 = *(const float4*)(ap + hi * 8);
            float4 f1 = *(const float4*)(ap + hi * 8 + 4);
            float4 f2 = *(const float4*)(ap + 16 + hi * 8);
            float4 f3 = *(const float4*)(ap + 16 + hi * 8 + 4);
            frag16& a = mrow ? a1 : a0;
            a.v[0]=(_Float16)f0.x;  a.v[1]=(_Float16)f0.y;  a.v[2]=(_Float16)f0.z;  a.v[3]=(_Float16)f0.w;
            a.v[4]=(_Float16)f1.x;  a.v[5]=(_Float16)f1.y;  a.v[6]=(_Float16)f1.z;  a.v[7]=(_Float16)f1.w;
            a.v[8]=(_Float16)f2.x;  a.v[9]=(_Float16)f2.y;  a.v[10]=(_Float16)f2.z; a.v[11]=(_Float16)f2.w;
            a.v[12]=(_Float16)f3.x; a.v[13]=(_Float16)f3.y; a.v[14]=(_Float16)f3.z; a.v[15]=(_Float16)f3.w;
        }
        #pragma unroll
        for (int t = 0; t < OUTC / 16; ++t) {
            const _Float16* wp = wrel + (long)(t * 16 + r) * HID + kk * 32;
            frag16 b;
            b.h[0] = *(const v8h*)(wp + hi * 8);
            b.h[1] = *(const v8h*)(wp + 16 + hi * 8);
            acc0[t] = __builtin_amdgcn_wmma_f32_16x16x32_f16(
                false, a0.v, false, b.v, (short)0, acc0[t], false, false);
            acc1[t] = __builtin_amdgcn_wmma_f32_16x16x32_f16(
                false, a1.v, false, b.v, (short)0, acc1[t], false, false);
        }
    }

    // ---- pass 1: root term, A from f16 activations (pure b128 loads) ----
    for (int kk = 0; kk < HID / 32; ++kk) {
        frag16 a0, a1;
        const _Float16* ap0 = hroot + (long)(tile * 32 + r) * HID + kk * 32;
        const _Float16* ap1 = ap0 + 16 * HID;
        a0.h[0] = *(const v8h*)(ap0 + hi * 8);
        a0.h[1] = *(const v8h*)(ap0 + 16 + hi * 8);
        a1.h[0] = *(const v8h*)(ap1 + hi * 8);
        a1.h[1] = *(const v8h*)(ap1 + 16 + hi * 8);
        #pragma unroll
        for (int t = 0; t < OUTC / 16; ++t) {
            const _Float16* wp = wroot + (long)(t * 16 + r) * HID + kk * 32;
            frag16 b;
            b.h[0] = *(const v8h*)(wp + hi * 8);
            b.h[1] = *(const v8h*)(wp + 16 + hi * 8);
            acc0[t] = __builtin_amdgcn_wmma_f32_16x16x32_f16(
                false, a0.v, false, b.v, (short)0, acc0[t], false, false);
            acc1[t] = __builtin_amdgcn_wmma_f32_16x16x32_f16(
                false, a1.v, false, b.v, (short)0, acc1[t], false, false);
        }
    }

    #pragma unroll
    for (int t = 0; t < OUTC / 16; ++t) {
        const int n = t * 16 + r;
        const float bn = brel[n];
        #pragma unroll
        for (int i = 0; i < 8; ++i) {
            int row0 = tile * 32 + i + 8 * hi;
            float v0 = acc0[t][i] + bn;
            float v1 = acc1[t][i] + bn;
            v0 = v0 > 0.f ? v0 : SLOPE * v0;
            v1 = v1 > 0.f ? v1 : SLOPE * v1;
            out[(long)row0 * OUTC + n]        = (OT)v0;
            out[(long)(row0 + 16) * OUTC + n] = (OT)v1;
        }
    }
}

// ============================================================
// Tail: LayerNorm(64) + fc2 (64->2); one node per thread.
// ============================================================
__global__ __launch_bounds__(256) void k_ln2_fc2(
    const float* __restrict__ t, const float* __restrict__ g,
    const float* __restrict__ b, const float* __restrict__ w,
    const float* __restrict__ bias, float* __restrict__ out)
{
    long i = (long)blockIdx.x * blockDim.x + threadIdx.x;
    if (i >= N_NODES) return;
    const float* ti = t + i * OUTF;

    float m = 0.f;
    #pragma unroll
    for (int k = 0; k < OUTF; ++k) m += ti[k];
    m *= (1.0f / OUTF);
    float v = 0.f;
    #pragma unroll
    for (int k = 0; k < OUTF; ++k) { float d = ti[k] - m; v += d * d; }
    v *= (1.0f / OUTF);
    const float rs = rsqrtf(v + LN_EPS);

    #pragma unroll
    for (int c = 0; c < NCLS; ++c) {
        float s = bias[c];
        #pragma unroll
        for (int k = 0; k < OUTF; ++k)
            s += ((ti[k] - m) * rs * g[k] + b[k]) * w[c * OUTF + k];
        out[i * NCLS + c] = s;
    }
}

// ============================================================
extern "C" void kernel_launch(void* const* d_in, const int* in_sizes, int n_in,
                              void* d_out, int out_size, void* d_ws, size_t ws_size,
                              hipStream_t stream)
{
    const float* x        = (const float*)d_in[0];
    const int*   ei       = (const int*)d_in[1];
    const float* ea       = (const float*)d_in[2];
    const float* ln1_g    = (const float*)d_in[3];
    const float* ln1_b    = (const float*)d_in[4];
    const float* fc1_w    = (const float*)d_in[5];
    const float* fc1_b    = (const float*)d_in[6];
    const float* c1_wrel  = (const float*)d_in[7];
    const float* c1_brel  = (const float*)d_in[8];
    const float* c1_wroot = (const float*)d_in[9];
    const float* c2_wrel  = (const float*)d_in[10];
    const float* c2_brel  = (const float*)d_in[11];
    const float* c2_wroot = (const float*)d_in[12];
    const float* ln2_g    = (const float*)d_in[13];
    const float* ln2_b    = (const float*)d_in[14];
    const float* fc2_w    = (const float*)d_in[15];
    const float* fc2_b    = (const float*)d_in[16];
    float* out = (float*)d_out;

    // ---- workspace layout ----
    // agg f32 [N,128] | t2 f32 [N,64] | h0 f16 [N,128] | h1 f16 [N,128] | weights f16
    float*     agg = (float*)d_ws;
    float*     t2  = agg + (size_t)N_NODES * HID;
    _Float16*  h0  = (_Float16*)(t2 + (size_t)N_NODES * OUTF);
    _Float16*  h1  = h0 + (size_t)N_NODES * HID;
    _Float16*  wf  = h1 + (size_t)N_NODES * HID;
    _Float16* fc1h    = wf;                      // 128*384
    _Float16* c1relh  = fc1h    + HID * IN_F;    // 128*128
    _Float16* c1rooth = c1relh  + HID * HID;     // 128*128
    _Float16* c2relh  = c1rooth + HID * HID;     // 64*128
    _Float16* c2rooth = c2relh  + OUTF * HID;    // 64*128

    const int ntiles = N_NODES / 32;             // 3125
    const long aggN  = (long)N_NODES * HID;
    const int zgrid  = (int)((aggN + 255) / 256);
    const int sgrid  = (int)(((long)E_EDGES * 32 + 255) / 256);
    auto cg = [](int n) { return (n + 255) / 256; };

    // 0) pre-convert weights to f16 (B fragments become pure b128 loads)
    k_cvt<<<cg(HID * IN_F),  256, 0, stream>>>(fc1_w,    fc1h,    HID * IN_F);
    k_cvt<<<cg(HID * HID),   256, 0, stream>>>(c1_wrel,  c1relh,  HID * HID);
    k_cvt<<<cg(HID * HID),   256, 0, stream>>>(c1_wroot, c1rooth, HID * HID);
    k_cvt<<<cg(OUTF * HID),  256, 0, stream>>>(c2_wrel,  c2relh,  OUTF * HID);
    k_cvt<<<cg(OUTF * HID),  256, 0, stream>>>(c2_wroot, c2rooth, OUTF * HID);

    // 1) LN1 + fc1 -> h0 (f16)
    k_ln_fc1<<<ntiles, 32, 0, stream>>>(x, ln1_g, ln1_b, fc1h, fc1_b, h0);

    // 2) conv1: scatter + dual GEMM + leaky -> h1 (f16)
    k_zero<<<zgrid, 256, 0, stream>>>(agg, aggN);
    k_scatter<<<sgrid, 256, 0, stream>>>(ei, ea, h0, agg);
    k_conv<HID, _Float16><<<ntiles, 32, 0, stream>>>(
        agg, h0, c1relh, c1_brel, c1rooth, h1);

    // 3) conv2: scatter + dual GEMM + leaky -> t2 (f32)
    k_zero<<<zgrid, 256, 0, stream>>>(agg, aggN);
    k_scatter<<<sgrid, 256, 0, stream>>>(ei, ea, h1, agg);
    k_conv<OUTF, float><<<ntiles, 32, 0, stream>>>(
        agg, h1, c2relh, c2_brel, c2rooth, t2);

    // 4) LN2 + fc2 -> out [N,2]
    k_ln2_fc2<<<(N_NODES + 255) / 256, 256, 0, stream>>>(
        t2, ln2_g, ln2_b, fc2_w, fc2_b, out);
}
